// GraphSAGERegressor_37847251812924
// MI455X (gfx1250) — matrix operand order repaired and checked by reading, not compiled, not run on hardware
//
#include <hip/hip_runtime.h>

// ---------------------------------------------------------------------------
// GraphSAGE (2x SAGEConv mean-aggr + linear head) for MI455X / gfx1250.
//   sparse:  bf16 gather + fp32 hardware atomics (L2-resident agg buffer)
//   dense:   v_wmma_f32_16x16x32_bf16, one wave per 16x16 output tile
// ---------------------------------------------------------------------------

typedef __attribute__((ext_vector_type(16))) __bf16 v16bf;
typedef __attribute__((ext_vector_type(8)))  __bf16 v8bf;
typedef __attribute__((ext_vector_type(8)))  float  v8f;

constexpr int N_NODES = 50000;
constexpr int N_EDGES = 800000;
constexpr int IN_DIM  = 128;
constexpr int HID     = 256;
constexpr int OUTD    = 13;

#define DEV static __device__ __forceinline__

DEV unsigned short f2bf(float f) {          // fp32 -> bf16, round-nearest-even
  unsigned u = __float_as_uint(f);
  u += 0x7FFFu + ((u >> 16) & 1u);
  return (unsigned short)(u >> 16);
}
DEV float bf2f(unsigned short h) { return __uint_as_float(((unsigned)h) << 16); }

// -------------------------------- conversion -------------------------------
__global__ void cvt_f32_bf16_kernel(const float* __restrict__ in,
                                    unsigned short* __restrict__ out, int n) {
  int i = blockIdx.x * blockDim.x + threadIdx.x;
  if (i < n) out[i] = f2bf(in[i]);
}

// -------------------------------- degree counts ----------------------------
__global__ void count_kernel(const int* __restrict__ dst, unsigned* __restrict__ cnt) {
  int e = blockIdx.x * blockDim.x + threadIdx.x;
  if (e < N_EDGES) atomicAdd(&cnt[dst[e]], 1u);
}
__global__ void inv_kernel(const unsigned* __restrict__ cnt_in, float* __restrict__ inv_out) {
  int i = blockIdx.x * blockDim.x + threadIdx.x;
  if (i < N_NODES) {
    float c = (float)cnt_in[i];
    inv_out[i] = 1.0f / fmaxf(c, 1.0f);
  }
}

// ------------------------- gather + scatter-add (mean aggr) ----------------
// One thread = one edge x 4 contiguous features. 8-byte bf16 load, four
// hardware global_atomic_add_f32 (no-return) into the fp32 agg buffer.
__global__ void scatter_kernel(const unsigned short* __restrict__ feat,
                               const int* __restrict__ src,
                               const int* __restrict__ dst,
                               float* __restrict__ agg,
                               int chunkShift /* log2(D/4) */) {
  int tid = blockIdx.x * blockDim.x + threadIdx.x;
  int e   = tid >> chunkShift;
  if (e >= N_EDGES) return;
  int ch = tid & ((1 << chunkShift) - 1);
  int D  = 4 << chunkShift;
  int s = src[e], d = dst[e];
  uint2 raw = *(const uint2*)(feat + (size_t)s * D + ch * 4);
  float* q  = agg + (size_t)d * D + ch * 4;
  unsafeAtomicAdd(q + 0, bf2f((unsigned short)(raw.x & 0xFFFFu)));
  unsafeAtomicAdd(q + 1, bf2f((unsigned short)(raw.x >> 16)));
  unsafeAtomicAdd(q + 2, bf2f((unsigned short)(raw.y & 0xFFFFu)));
  unsafeAtomicAdd(q + 3, bf2f((unsigned short)(raw.y >> 16)));
}

// ------------------------------ mean finalize ------------------------------
__global__ void mean_kernel(const float* __restrict__ agg,
                            const float* __restrict__ inv,
                            unsigned short* __restrict__ out,
                            int n, int dShift /* log2(D) */) {
  int i = blockIdx.x * blockDim.x + threadIdx.x;
  if (i < n) out[i] = f2bf(agg[i] * inv[i >> dShift]);
}

// ------------------------------ WMMA fragments -----------------------------
// A (16x32 bf16): lane L -> row m=L&15, hi=L>>4.
//   elements 0..15 = K k0+16*hi .. k0+16*hi+15 contiguous
DEV v16bf load_a_frag(const unsigned short* __restrict__ row, int k0, int hi) {
  v8bf lo = *(const v8bf*)(row + k0 + 16 * hi);
  v8bf hh = *(const v8bf*)(row + k0 + 16 * hi + 8);
  return __builtin_shufflevector(lo, hh, 0,1,2,3,4,5,6,7,8,9,10,11,12,13,14,15);
}
// B (32x16 bf16): lane L -> col n=L&15; K = k0+16*hi .. k0+16*hi+15 contiguous.
// (out = in @ W.T  =>  B[k][n] = W[n][k], W row-major => contiguous per lane)
DEV v16bf load_b_frag(const unsigned short* __restrict__ wrow, int k0, int hi) {
  v8bf lo = *(const v8bf*)(wrow + k0 + 16 * hi);
  v8bf hh = *(const v8bf*)(wrow + k0 + 16 * hi + 8);
  return __builtin_shufflevector(lo, hh, 0,1,2,3,4,5,6,7,8,9,10,11,12,13,14,15);
}

// ------------------- fused SAGE layer: relu(m@Wl.T + b + x@Wr.T) -----------
// One wave per 16x16 output tile; K-loop in steps of 32 with bf16 WMMA.
template <int K>
__global__ void sage_gemm_kernel(const unsigned short* __restrict__ a1, // mean  [N,K]
                                 const unsigned short* __restrict__ a2, // x/h   [N,K]
                                 const unsigned short* __restrict__ w1, // Wl [HID,K]
                                 const unsigned short* __restrict__ w2, // Wr [HID,K]
                                 const float* __restrict__ bias,        // [HID]
                                 unsigned short* __restrict__ out)      // [N,HID]
{
  constexpr int NT = HID / 16;                                   // 16 n-tiles
  int wave  = (blockIdx.x * blockDim.x + threadIdx.x) >> 5;      // uniform/wave
  int lane  = threadIdx.x & 31;
  int mtile = wave / NT, ntile = wave % NT;
  int ln = lane & 15, hi = lane >> 4;
  int nglob = ntile * 16 + ln;

  const unsigned short* ar1 = a1 + (size_t)(mtile * 16 + ln) * K;
  const unsigned short* ar2 = a2 + (size_t)(mtile * 16 + ln) * K;
  const unsigned short* wr1 = w1 + (size_t)nglob * K;
  const unsigned short* wr2 = w2 + (size_t)nglob * K;

  v8f c = {};
#pragma unroll
  for (int k0 = 0; k0 < K; k0 += 32) {
    v16bf A = load_a_frag(ar1, k0, hi);
    v16bf B = load_b_frag(wr1, k0, hi);
    c = __builtin_amdgcn_wmma_f32_16x16x32_bf16(false, A, false, B, (short)0, c, false, false);
  }
#pragma unroll
  for (int k0 = 0; k0 < K; k0 += 32) {
    v16bf A = load_a_frag(ar2, k0, hi);
    v16bf B = load_b_frag(wr2, k0, hi);
    c = __builtin_amdgcn_wmma_f32_16x16x32_bf16(false, A, false, B, (short)0, c, false, false);
  }

  // C/D layout: n = lane&15, VGPR r -> m = r + 8*hi. Fused bias + ReLU + bf16.
  float bv = bias[nglob];
  int mbase = mtile * 16 + 8 * hi;
#pragma unroll
  for (int r = 0; r < 8; ++r) {
    float v = fmaxf(c[r] + bv, 0.0f);
    out[(size_t)(mbase + r) * HID + nglob] = f2bf(v);
  }
}

// ---------------------- final linear head: h2 @ Wlin.T + b -----------------
// Wlin padded to 16x256 (zero rows 13..15); stores mask n < 13, fp32 output.
__global__ void final_gemm_kernel(const unsigned short* __restrict__ a,   // [N,HID]
                                  const unsigned short* __restrict__ w,   // [16,HID]
                                  const float* __restrict__ bias,         // [13]
                                  float* __restrict__ out)                // [N,13]
{
  int wave = (blockIdx.x * blockDim.x + threadIdx.x) >> 5;
  if (wave >= N_NODES / 16) return;   // wave-uniform guard -> EXEC all ones
  int lane = threadIdx.x & 31;
  int ln = lane & 15, hi = lane >> 4;

  const unsigned short* ar = a + (size_t)(wave * 16 + ln) * HID;
  const unsigned short* wr = w + (size_t)ln * HID;

  v8f c = {};
#pragma unroll
  for (int k0 = 0; k0 < HID; k0 += 32) {
    v16bf A = load_a_frag(ar, k0, hi);
    v16bf B = load_b_frag(wr, k0, hi);
    c = __builtin_amdgcn_wmma_f32_16x16x32_bf16(false, A, false, B, (short)0, c, false, false);
  }

  if (ln < OUTD) {
    float bv = bias[ln];
    int mbase = wave * 16 + 8 * hi;
#pragma unroll
    for (int r = 0; r < 8; ++r)
      out[(size_t)(mbase + r) * OUTD + ln] = c[r] + bv;
  }
}

// ---------------------------------------------------------------------------
extern "C" void kernel_launch(void* const* d_in, const int* in_sizes, int n_in,
                              void* d_out, int out_size, void* d_ws, size_t ws_size,
                              hipStream_t stream) {
  const float* x     = (const float*)d_in[0];
  const int*   ei    = (const int*)d_in[1];
  const int*   src   = ei;
  const int*   dst   = ei + N_EDGES;
  const float* w1_l  = (const float*)d_in[2];
  const float* b1_l  = (const float*)d_in[3];
  const float* w1_r  = (const float*)d_in[4];
  const float* w2_l  = (const float*)d_in[5];
  const float* b2_l  = (const float*)d_in[6];
  const float* w2_r  = (const float*)d_in[7];
  const float* w_lin = (const float*)d_in[8];
  const float* b_lin = (const float*)d_in[9];
  float* out = (float*)d_out;

  // ---- workspace carve-out (~141 MB total) ----
  char*  ws  = (char*)d_ws;
  size_t off = 0;
  auto take = [&](size_t bytes) -> char* {
    char* p = ws + off;
    off = (off + bytes + 255) & ~(size_t)255;
    return p;
  };
  unsigned short* xbf    = (unsigned short*)take((size_t)N_NODES * IN_DIM * 2);
  unsigned short* meanbf = (unsigned short*)take((size_t)N_NODES * HID * 2);
  unsigned short* h1bf   = (unsigned short*)take((size_t)N_NODES * HID * 2);
  unsigned short* h2bf   = (unsigned short*)take((size_t)N_NODES * HID * 2);
  unsigned short* w1lbf  = (unsigned short*)take((size_t)HID * IN_DIM * 2);
  unsigned short* w1rbf  = (unsigned short*)take((size_t)HID * IN_DIM * 2);
  unsigned short* w2lbf  = (unsigned short*)take((size_t)HID * HID * 2);
  unsigned short* w2rbf  = (unsigned short*)take((size_t)HID * HID * 2);
  unsigned short* wlinbf = (unsigned short*)take((size_t)16 * HID * 2);
  void*           cntbuf = (void*)take((size_t)N_NODES * 4);       // u32 then f32 in place
  float*          agg    = (float*)take((size_t)N_NODES * HID * 4);

  const int T = 256;

  // zero accumulators / padded weight (capture-safe async memsets)
  hipMemsetAsync(cntbuf, 0, (size_t)N_NODES * 4, stream);
  hipMemsetAsync(agg,    0, (size_t)N_NODES * IN_DIM * 4, stream);
  hipMemsetAsync(wlinbf, 0, (size_t)16 * HID * 2, stream);

  // fp32 -> bf16 conversions (activations once, weights once per call)
  cvt_f32_bf16_kernel<<<(N_NODES * IN_DIM + T - 1) / T, T, 0, stream>>>(x, xbf, N_NODES * IN_DIM);
  cvt_f32_bf16_kernel<<<(HID * IN_DIM + T - 1) / T, T, 0, stream>>>(w1_l, w1lbf, HID * IN_DIM);
  cvt_f32_bf16_kernel<<<(HID * IN_DIM + T - 1) / T, T, 0, stream>>>(w1_r, w1rbf, HID * IN_DIM);
  cvt_f32_bf16_kernel<<<(HID * HID + T - 1) / T, T, 0, stream>>>(w2_l, w2lbf, HID * HID);
  cvt_f32_bf16_kernel<<<(HID * HID + T - 1) / T, T, 0, stream>>>(w2_r, w2rbf, HID * HID);
  cvt_f32_bf16_kernel<<<(OUTD * HID + T - 1) / T, T, 0, stream>>>(w_lin, wlinbf, OUTD * HID);

  // degree counts -> 1/max(cnt,1)
  count_kernel<<<(N_EDGES + T - 1) / T, T, 0, stream>>>(dst, (unsigned*)cntbuf);
  inv_kernel<<<(N_NODES + T - 1) / T, T, 0, stream>>>((const unsigned*)cntbuf, (float*)cntbuf);

  // -------- layer 1 --------
  scatter_kernel<<<(N_EDGES * (IN_DIM / 4) + T - 1) / T, T, 0, stream>>>(xbf, src, dst, agg, 5);
  mean_kernel<<<(N_NODES * IN_DIM + T - 1) / T, T, 0, stream>>>(agg, (const float*)cntbuf, meanbf,
                                                                N_NODES * IN_DIM, 7);
  // 3125 m-tiles * 16 n-tiles = 50000 waves, 8 waves/block
  sage_gemm_kernel<IN_DIM><<<(N_NODES / 16) * (HID / 16) / 8, T, 0, stream>>>(
      meanbf, xbf, w1lbf, w1rbf, b1_l, h1bf);

  // -------- layer 2 --------
  hipMemsetAsync(agg, 0, (size_t)N_NODES * HID * 4, stream);
  scatter_kernel<<<(N_EDGES * (HID / 4) + T - 1) / T, T, 0, stream>>>(h1bf, src, dst, agg, 6);
  mean_kernel<<<(N_NODES * HID + T - 1) / T, T, 0, stream>>>(agg, (const float*)cntbuf, meanbf,
                                                             N_NODES * HID, 8);
  sage_gemm_kernel<HID><<<(N_NODES / 16) * (HID / 16) / 8, T, 0, stream>>>(
      meanbf, h1bf, w2lbf, w2rbf, b2_l, h2bf);

  // -------- linear head --------
  final_gemm_kernel<<<((N_NODES / 16) + 7) / 8, T, 0, stream>>>(h2bf, wlinbf, b_lin, out);
}
